// cross_scale_attention_spe_80874234184228
// MI455X (gfx1250) — matrix operand order repaired and checked by reading, not compiled, not run on hardware
//
#include <hip/hip_runtime.h>
#include <hip/hip_bf16.h>

typedef __attribute__((ext_vector_type(16))) _Float16 v16h;
typedef __attribute__((ext_vector_type(8)))  float    v8f;

__device__ __forceinline__ float lrelu(float x) { return x >= 0.f ? x : 0.01f * x; }

// ---------------------------------------------------------------------------
// K1: v = lrelu(conv3x3(hrms, Wv, bv, stride 1, pad 1)) -> f16
// one thread per output element (n,i,y,x); x fastest -> coalesced
// ---------------------------------------------------------------------------
__global__ __launch_bounds__(256)
void conv_v_kernel(const float* __restrict__ hrms, const float* __restrict__ Wv,
                   const float* __restrict__ bv, _Float16* __restrict__ vout) {
    __shared__ float w[8 * 8 * 9];
    __shared__ float bb[8];
    for (int idx = threadIdx.x; idx < 576; idx += 256) w[idx] = Wv[idx];
    if (threadIdx.x < 8) bb[threadIdx.x] = bv[threadIdx.x];
    __syncthreads();

    int id = blockIdx.x * 256 + threadIdx.x;     // 0 .. 2*8*512*512-1
    int x = id & 511;
    int y = (id >> 9) & 511;
    int i = (id >> 18) & 7;
    int n = id >> 21;

    float acc = bb[i];
    for (int ci = 0; ci < 8; ++ci) {
        const float* hp = hrms + (((n * 8 + ci) * 512) << 9);
        const float* wp = w + (i * 8 + ci) * 9;
        for (int ky = 0; ky < 3; ++ky) {
            int yy = y + ky - 1;
            if (yy < 0 || yy >= 512) continue;
            for (int kx = 0; kx < 3; ++kx) {
                int xx = x + kx - 1;
                if (xx < 0 || xx >= 512) continue;
                acc += wp[ky * 3 + kx] * hp[(yy << 9) + xx];
            }
        }
    }
    vout[id] = (_Float16)lrelu(acc);
}

// ---------------------------------------------------------------------------
// K2: q = lrelu(conv3x3(msi, Wq, bq, stride 16, pad 1)) -> (2,8,32,32) f32
// ---------------------------------------------------------------------------
__global__ __launch_bounds__(256)
void conv_q_kernel(const float* __restrict__ msi, const float* __restrict__ Wq,
                   const float* __restrict__ bq, float* __restrict__ qout) {
    __shared__ float w[8 * 8 * 9];
    __shared__ float bb[8];
    for (int idx = threadIdx.x; idx < 576; idx += 256) w[idx] = Wq[idx];
    if (threadIdx.x < 8) bb[threadIdx.x] = bq[threadIdx.x];
    __syncthreads();

    int id = blockIdx.x * 256 + threadIdx.x;     // 0 .. 2*8*32*32-1
    int ox = id & 31;
    int oy = (id >> 5) & 31;
    int i  = (id >> 10) & 7;
    int n  = id >> 13;

    float acc = bb[i];
    for (int ci = 0; ci < 8; ++ci) {
        const float* hp = msi + (((n * 8 + ci) * 512) << 9);
        const float* wp = w + (i * 8 + ci) * 9;
        for (int ky = 0; ky < 3; ++ky) {
            int iy = 16 * oy - 1 + ky;
            if (iy < 0 || iy >= 512) continue;
            for (int kx = 0; kx < 3; ++kx) {
                int ix = 16 * ox - 1 + kx;
                if (ix < 0 || ix >= 512) continue;
                acc += wp[ky * 3 + kx] * hp[(iy << 9) + ix];
            }
        }
    }
    qout[id] = lrelu(acc);
}

// ---------------------------------------------------------------------------
// K3: k = lrelu(conv3x3(hsi, Wk, bk, stride 4, pad 1)) -> (2,128,32,32) f32
// block = one (n, cout); Wk row staged in LDS; each thread does 4 pixels
// ---------------------------------------------------------------------------
__global__ __launch_bounds__(256)
void conv_k_kernel(const float* __restrict__ hsi, const float* __restrict__ Wk,
                   const float* __restrict__ bk, float* __restrict__ kout) {
    __shared__ float wk[1152];
    int b = blockIdx.x;
    int n = b >> 7;
    int c = b & 127;
    for (int idx = threadIdx.x; idx < 1152; idx += 256) wk[idx] = Wk[c * 1152 + idx];
    __syncthreads();

    float bias = bk[c];
    for (int pp = 0; pp < 4; ++pp) {
        int p  = pp * 256 + threadIdx.x;         // 0..1023
        int oy = p >> 5;
        int ox = p & 31;
        float acc = bias;
        for (int ci = 0; ci < 128; ++ci) {
            const float* hp = hsi + (((n * 128 + ci) * 128) << 7);
            const float* wp = wk + ci * 9;
            for (int ky = 0; ky < 3; ++ky) {
                int iy = 4 * oy - 1 + ky;
                if (iy < 0 || iy >= 128) continue;
                for (int kx = 0; kx < 3; ++kx) {
                    int ix = 4 * ox - 1 + kx;
                    if (ix < 0 || ix >= 128) continue;
                    acc += wp[ky * 3 + kx] * hp[(iy << 7) + ix];
                }
            }
        }
        kout[(n * 128 + c) * 1024 + p] = lrelu(acc);
    }
}

// ---------------------------------------------------------------------------
// K4: k0max[n] = max_i ||q[n,i]||_2 ; one block per sample
// ---------------------------------------------------------------------------
__global__ __launch_bounds__(256)
void norm_kernel(const float* __restrict__ q, float* __restrict__ k0) {
    __shared__ float red[256];
    __shared__ float mx;
    int n = blockIdx.x;
    if (threadIdx.x == 0) mx = 0.f;
    __syncthreads();
    for (int i = 0; i < 8; ++i) {
        const float* qp = q + (n * 8 + i) * 1024;
        float s = 0.f;
        for (int p = threadIdx.x; p < 1024; p += 256) { float v = qp[p]; s += v * v; }
        red[threadIdx.x] = s;
        __syncthreads();
        for (int off = 128; off > 0; off >>= 1) {
            if (threadIdx.x < off) red[threadIdx.x] += red[threadIdx.x + off];
            __syncthreads();
        }
        if (threadIdx.x == 0) { float nv = sqrtf(red[0]); if (nv > mx) mx = nv; }
        __syncthreads();
    }
    if (threadIdx.x == 0) k0[n] = mx;
}

// ---------------------------------------------------------------------------
// K5: att[n,i,c] = softmax_c( 10 * <q[n,i]/k0max, k[n,c]> ) ; block = (n,i), thread = c
// ---------------------------------------------------------------------------
__global__ __launch_bounds__(128)
void att_kernel(const float* __restrict__ q, const float* __restrict__ kbuf,
                const float* __restrict__ k0, float* __restrict__ att) {
    __shared__ float qrow[1024];
    __shared__ float red[128];
    int b = blockIdx.x;
    int n = b >> 3;
    int i = b & 7;
    int c = threadIdx.x;
    const float* qp = q + (n * 8 + i) * 1024;
    for (int p = c; p < 1024; p += 128) qrow[p] = qp[p];
    __syncthreads();

    const float* kp = kbuf + (n * 128 + c) * 1024;
    float dot = 0.f;
    for (int p = 0; p < 1024; ++p) dot += qrow[p] * kp[p];
    float a = dot * 10.0f / k0[n];

    red[c] = a;
    __syncthreads();
    for (int off = 64; off > 0; off >>= 1) {
        if (c < off) red[c] = fmaxf(red[c], red[c + off]);
        __syncthreads();
    }
    float amax = red[0];
    __syncthreads();
    float e = __expf(a - amax);
    red[c] = e;
    __syncthreads();
    for (int off = 64; off > 0; off >>= 1) {
        if (c < off) red[c] += red[c + off];
        __syncthreads();
    }
    att[(n * 8 + i) * 128 + c] = e / red[0];
}

// ---------------------------------------------------------------------------
// K6: W_eff, stored SWIZZLED for direct WMMA B-fragment loads:
//   weff[n][kc][half][o][j]  with  k = kc*32 + half*16 + j,
//   value = (1/6) * sum_c Wr[o,c,d] * att[n,i,c], k = i*9+d, zeros for k>=72.
// Each lane of the main kernel then reads its 16 B elements as one contiguous
// 32-byte LDS vector. block = (n,o), thread = k (96)
// ---------------------------------------------------------------------------
__global__ __launch_bounds__(96)
void weff_kernel(const float* __restrict__ Wr, const float* __restrict__ att,
                 _Float16* __restrict__ weff) {
    __shared__ float at[1024];
    int b = blockIdx.x;
    int n = b >> 7;
    int o = b & 127;
    int k = threadIdx.x;                         // 0..95
    for (int idx = k; idx < 1024; idx += 96) at[idx] = att[n * 1024 + idx];
    __syncthreads();

    _Float16 val = (_Float16)0.f;
    if (k < 72) {
        int i = k / 9;
        int d = k - i * 9;
        const float* wr = Wr + o * 128 * 9 + d;  // Wr[o][c][d], stride 9 over c
        float s = 0.f;
        for (int c = 0; c < 128; ++c) s += wr[c * 9] * at[i * 128 + c];
        val = (_Float16)(s * (1.0f / 6.0f));
    }
    int kc   = k >> 5;                           // 0..2
    int rem  = k & 31;
    int half = rem >> 4;                         // 0..1
    int j    = rem & 15;                         // 0..15
    // [n][kc][half][o][j]
    weff[((((n * 3 + kc) * 2 + half) * 128 + o) * 16) + j] = val;
}

// ---------------------------------------------------------------------------
// K7: fused mix + conv_r via WMMA implicit GEMM.
// block = (n, y, 128-pixel x strip). M=128 px (8 waves x 16), N=128 oc, K=96.
// A = im2col(v_f16) from LDS halo; B = swizzled W_eff panel in LDS (one
// aligned 32B vector load per fragment); D staged through LDS in 4 phases of
// 32 oc for coalesced float4 stores with bias + lrelu.
// ---------------------------------------------------------------------------
__global__ __launch_bounds__(256)
void fused_convr_wmma_kernel(const _Float16* __restrict__ vh,
                             const _Float16* __restrict__ weff,
                             const float* __restrict__ br,
                             float* __restrict__ out) {
    __shared__ __attribute__((aligned(32))) _Float16 ldsV[8 * 3 * 132]; // halo
    __shared__ __attribute__((aligned(32))) _Float16 ldsB[96 * 128];    // swizzled B
    __shared__ __attribute__((aligned(16))) float    ldsD[32 * 132];    // 32 oc x 128 px

    const int tid = threadIdx.x;
    const int b   = blockIdx.x;
    const int n   = b >> 11;                  // 2048 blocks per sample
    const int rem = b & 2047;
    const int y   = rem >> 2;
    const int x0  = (rem & 3) << 7;           // 0,128,256,384

    // stage v halo: rows y-1..y+1, x0-1..x0+128, zeros on OOB
    for (int idx = tid; idx < 8 * 3 * 132; idx += 256) {
        int row = idx / 132;
        int xv  = idx - row * 132;            // 0..131 (130/131 padding)
        int i   = row / 3;
        int r   = row - i * 3;
        int yy  = y + r - 1;
        int xx  = x0 - 1 + xv;
        _Float16 val = (_Float16)0.f;
        if (yy >= 0 && yy < 512 && xx >= 0 && xx < 512 && xv < 130)
            val = vh[(((n * 8 + i) * 512 + yy) << 9) + xx];
        ldsV[idx] = val;
    }
    // stage swizzled B panel: verbatim 24576-byte copy as uint4 (b128 ops)
    {
        const uint4* src = (const uint4*)(weff + n * 96 * 128);
        uint4* dst = (uint4*)ldsB;
        for (int idx = tid; idx < (96 * 128 * 2) / 16; idx += 256)  // 1536 uint4s
            dst[idx] = src[idx];
    }
    __syncthreads();

    const int lane = tid & 31;
    const int w    = tid >> 5;                // wave id: pixel group
    const int half = lane >> 4;
    const int m    = lane & 15;
    const int xx0  = w * 16 + m;              // tap x-index base in ldsV (+kx)

    // Build the 3 A fragments (contiguous K order, same as B: lanes<16 hold K 0-15,
    // lanes>=16 hold K 16-31), K>=72 zero-padded.  One-time gather.
    v16h af[3];
#pragma unroll
    for (int kc = 0; kc < 3; ++kc) {
#pragma unroll
        for (int j = 0; j < 16; ++j) {
            int K = kc * 32 + half * 16 + j;
            _Float16 val = (_Float16)0.f;
            if (K < 72) {
                int i  = K / 9;
                int d  = K - i * 9;
                int ky = d / 3;
                int kx = d - ky * 3;
                val = ldsV[(i * 3 + ky) * 132 + xx0 + kx];
            }
            af[kc][j] = val;
        }
    }

    for (int phase = 0; phase < 4; ++phase) {
#pragma unroll
        for (int t2 = 0; t2 < 2; ++t2) {
            int nt   = phase * 2 + t2;
            int ocol = nt * 16 + m;
            v8f acc = {};
#pragma unroll
            for (int kc = 0; kc < 3; ++kc) {
                // swizzled layout: one aligned 32-byte vector per fragment/lane
                const v16h* bp =
                    (const v16h*)&ldsB[((kc * 2 + half) * 128 + ocol) * 16];
                v16h bf = *bp;
                acc = __builtin_amdgcn_wmma_f32_16x16x32_f16(
                    false, af[kc], false, bf, (short)0, acc, false, false);
            }
            // dump D tile: VGPR r -> row M = r + half*8, col = lane&15
#pragma unroll
            for (int r = 0; r < 8; ++r) {
                int M = r + half * 8;
                ldsD[(t2 * 16 + m) * 132 + w * 16 + M] = acc[r];
            }
        }
        __syncthreads();
        // coalesced writeout of 32 oc x 128 px with bias + lrelu
        {
            int o_loc = tid >> 3;             // 0..31
            int seg   = tid & 7;              // 0..7 -> 16-px segment
            int o     = phase * 32 + o_loc;
            float bias = br[o];
            float* orow = out + ((((size_t)n * 128 + o) * 512 + y) << 9) + x0 + seg * 16;
            const float* drow = &ldsD[o_loc * 132 + seg * 16];
#pragma unroll
            for (int j = 0; j < 4; ++j) {
                float4 v4 = *(const float4*)(drow + j * 4);
                float4 res;
                res.x = lrelu(v4.x + bias);
                res.y = lrelu(v4.y + bias);
                res.z = lrelu(v4.z + bias);
                res.w = lrelu(v4.w + bias);
                *(float4*)(orow + j * 4) = res;
            }
        }
        __syncthreads();
    }
}

// ---------------------------------------------------------------------------
// launch
// ---------------------------------------------------------------------------
extern "C" void kernel_launch(void* const* d_in, const int* in_sizes, int n_in,
                              void* d_out, int out_size, void* d_ws, size_t ws_size,
                              hipStream_t stream) {
    const float* hrms = (const float*)d_in[0];
    const float* msi  = (const float*)d_in[1];
    const float* hsi  = (const float*)d_in[2];
    const float* Wv   = (const float*)d_in[3];
    const float* bv   = (const float*)d_in[4];
    const float* Wq   = (const float*)d_in[5];
    const float* bq   = (const float*)d_in[6];
    const float* Wk   = (const float*)d_in[7];
    const float* bk   = (const float*)d_in[8];
    const float* Wr   = (const float*)d_in[9];
    const float* br   = (const float*)d_in[10];
    float* outp = (float*)d_out;

    char* ws = (char*)d_ws;
    _Float16* vh   = (_Float16*)(ws);                 // 2*8*512*512 f16 = 8 MB
    float*    qb   = (float*)(ws + 8388608);          // 2*8*32*32 f32   = 64 KB
    float*    kb   = (float*)(ws + 8454144);          // 2*128*32*32 f32 = 1 MB
    float*    k0   = (float*)(ws + 9502720);          // 2 f32 (padded)
    float*    attb = (float*)(ws + 9502976);          // 2*8*128 f32     = 8 KB
    _Float16* weff = (_Float16*)(ws + 9511168);       // 2*96*128 f16    = 48 KB (swizzled)

    conv_v_kernel<<<16384, 256, 0, stream>>>(hrms, Wv, bv, vh);
    conv_q_kernel<<<64, 256, 0, stream>>>(msi, Wq, bq, qb);
    conv_k_kernel<<<256, 256, 0, stream>>>(hsi, Wk, bk, kb);
    norm_kernel<<<2, 256, 0, stream>>>(qb, k0);
    att_kernel<<<16, 128, 0, stream>>>(qb, kb, k0, attb);
    weff_kernel<<<256, 96, 0, stream>>>(Wr, attb, weff);
    fused_convr_wmma_kernel<<<4096, 256, 0, stream>>>(vh, weff, br, outp);
}
